// TransformerLayer_9534827397806
// MI455X (gfx1250) — matrix operand order repaired and checked by reading, not compiled, not run on hardware
//
#include <hip/hip_runtime.h>

#define N_NODES 4096
#define IN_DIM  128
#define KQ_DIM  64
#define V_DIM   128

typedef __attribute__((ext_vector_type(16))) _Float16 v16h;
typedef __attribute__((ext_vector_type(8)))  _Float16 v8h;
typedef __attribute__((ext_vector_type(8)))  float    v8f;
typedef __attribute__((ext_vector_type(4)))  float    f32x4;

// ---- A-operand fragment (16xK f16, row-major, K32 window at kofs) -----------
// ISA 7.12.2: lanes 0-15 rows M=0..15 hold K = kofs + {hgrp*8..+7, 16+hgrp*8..+7}
static __device__ __forceinline__ v16h load_a_frag(const _Float16* base,
                                                   int stride, int kofs, int lane) {
  const int m = lane & 15, hgrp = lane >> 4;
  const _Float16* p = base + (size_t)m * stride + kofs + hgrp * 8;
  v8h lo = *(const v8h*)(p);
  v8h hi = *(const v8h*)(p + 16);
  v16h a;
#pragma unroll
  for (int t = 0; t < 8; ++t) { a[t] = lo[t]; a[8 + t] = hi[t]; }
  return a;
}

// ---- B-operand fragment: B[k][n] = mat[n0+n][kbase+k] (K = {khalf*8..+7, 16+khalf*8..+7})
static __device__ __forceinline__ v16h load_b_frag(const _Float16* __restrict__ mat,
                                                   size_t stride, int n0, int kbase, int lane) {
  const int n = lane & 15, khalf = lane >> 4;
  const _Float16* p = mat + (size_t)(n0 + n) * stride + kbase + khalf * 8;
  v8h lo = *(const v8h*)(p);
  v8h hi = *(const v8h*)(p + 16);
  v16h b;
#pragma unroll
  for (int t = 0; t < 8; ++t) { b[t] = lo[t]; b[8 + t] = hi[t]; }
  return b;
}

// =============================================================================
// Kernel 0: transpose output-MLP weights to f16 [out][in] for B-operand loads
// =============================================================================
__global__ __launch_bounds__(256)
void prep_weights(const float* __restrict__ l_w1, const float* __restrict__ l_w2,
                  _Float16* __restrict__ w1T, _Float16* __restrict__ w2T) {
  const int idx = blockIdx.x * 256 + threadIdx.x;   // 0..16383
  const int d = idx >> 7, c = idx & 127;
  w1T[(size_t)c * IN_DIM + d] = (_Float16)l_w1[idx];
  w2T[(size_t)c * IN_DIM + d] = (_Float16)l_w2[idx];
}

// =============================================================================
// Kernel 1: q = (x@w_q)/8 -> f16 [N,64]; k = x@w_k -> f16 [N,64];
//           vpT = (x@w_v + b_v)^T -> f16 [128,N]  (transposed for B-operand)
// =============================================================================
__global__ __launch_bounds__(256)
void proj_kernel(const float* __restrict__ x,
                 const float* __restrict__ w_q,
                 const float* __restrict__ w_k,
                 const float* __restrict__ w_v,
                 const float* __restrict__ b_v,
                 _Float16* __restrict__ q16,
                 _Float16* __restrict__ k16,
                 _Float16* __restrict__ vpT)
{
  __shared__ float xs[16][IN_DIM];
  const int tid = threadIdx.x;
  const int i0  = blockIdx.x * 16;
  for (int idx = tid; idx < 16 * IN_DIM; idx += 256)
    xs[idx >> 7][idx & 127] = x[(size_t)(i0 + (idx >> 7)) * IN_DIM + (idx & 127)];
  __syncthreads();

  float acc[16];
#pragma unroll
  for (int r = 0; r < 16; ++r) acc[r] = 0.0f;

  if (tid < 64) {                       // Q columns
    const int c = tid;
    for (int d = 0; d < IN_DIM; ++d) {
      const float w = w_q[d * KQ_DIM + c];
#pragma unroll
      for (int r = 0; r < 16; ++r) acc[r] = fmaf(xs[r][d], w, acc[r]);
    }
#pragma unroll
    for (int r = 0; r < 16; ++r)
      q16[(size_t)(i0 + r) * KQ_DIM + c] = (_Float16)(acc[r] * 0.125f); // 1/sqrt(64)
  } else if (tid < 128) {               // K columns
    const int c = tid - 64;
    for (int d = 0; d < IN_DIM; ++d) {
      const float w = w_k[d * KQ_DIM + c];
#pragma unroll
      for (int r = 0; r < 16; ++r) acc[r] = fmaf(xs[r][d], w, acc[r]);
    }
#pragma unroll
    for (int r = 0; r < 16; ++r)
      k16[(size_t)(i0 + r) * KQ_DIM + c] = (_Float16)acc[r];
  } else {                              // V columns, stored transposed
    const int c = tid - 128;
    for (int d = 0; d < IN_DIM; ++d) {
      const float w = w_v[d * V_DIM + c];
#pragma unroll
      for (int r = 0; r < 16; ++r) acc[r] = fmaf(xs[r][d], w, acc[r]);
    }
    const float bv = b_v[c];
    v8h lo, hi;
#pragma unroll
    for (int r = 0; r < 8; ++r) {
      lo[r] = (_Float16)(acc[r] + bv);
      hi[r] = (_Float16)(acc[r + 8] + bv);
    }
    *(v8h*)(vpT + (size_t)c * N_NODES + i0)     = lo;
    *(v8h*)(vpT + (size_t)c * N_NODES + i0 + 8) = hi;
  }
}

// =============================================================================
// Kernel 2: fused edge-MLP + flash attention + residual + WMMA output MLP
// grid = 256 (one 16-row tile each), block = 256 threads = 8 waves.
// Wave w owns columns [w*512, w*512+512), 64 at a time; partials merged in LDS.
// =============================================================================
__global__ __launch_bounds__(256)
void attn_kernel(const float* __restrict__ x,
                 const float* __restrict__ edge_attr,
                 const _Float16* __restrict__ q16,
                 const _Float16* __restrict__ k16,
                 const _Float16* __restrict__ vpT,
                 const _Float16* __restrict__ w1T16,
                 const _Float16* __restrict__ w2T16,
                 const float* __restrict__ e_w1, const float* __restrict__ e_b1,
                 const float* __restrict__ e_w2, const float* __restrict__ e_b2,
                 const float* __restrict__ l_b1, const float* __restrict__ l_b2,
                 float* __restrict__ out)
{
  __shared__ _Float16 Plds[8][16 * 72];   // per-wave P tiles; reused by epilogue
  __shared__ float outbuf[16][V_DIM];
  __shared__ float stmax[8][16];
  __shared__ float stsum[8][16];
  __shared__ float vtile[16][V_DIM];

  const int tid  = threadIdx.x;
  const int lane = tid & 31;
  const int wave = tid >> 5;
  const int hgrp = lane >> 4;
  const int nlo  = lane & 15;
  const int i0   = blockIdx.x * 16;

  for (int idx = tid; idx < 16 * V_DIM; idx += 256)
    outbuf[idx >> 7][idx & 127] = 0.0f;

  // Edge-MLP weights: uniform addresses -> scalar loads
  float W1[4][8], B1[8], W2[8];
#pragma unroll
  for (int d = 0; d < 4; ++d)
#pragma unroll
    for (int h = 0; h < 8; ++h) W1[d][h] = e_w1[d * 8 + h];
#pragma unroll
  for (int h = 0; h < 8; ++h) { B1[h] = e_b1[h]; W2[h] = e_w2[h]; }
  const float B2 = e_b2[0];

  const v16h ak0 = load_a_frag(k16 + (size_t)i0 * KQ_DIM, KQ_DIM, 0,  lane);
  const v16h ak1 = load_a_frag(k16 + (size_t)i0 * KQ_DIM, KQ_DIM, 32, lane);

  v8f acc[8] = {};
  float rmax[8], rsum[8];
#pragma unroll
  for (int r = 0; r < 8; ++r) { rmax[r] = -__builtin_inff(); rsum[r] = 0.0f; }

  const int jbeg = wave * 512;
  for (int j0 = jbeg; j0 < jbeg + 512; j0 += 64) {
    if (j0 + 64 < jbeg + 512) {                 // prefetch next 16x64 edge tile
#pragma unroll
      for (int t = 0; t < 2; ++t) {
        const int L = lane + t * 32;
        __builtin_prefetch(edge_attr +
            ((size_t)(i0 + (L >> 2)) * N_NODES + (j0 + 64) + (L & 3) * 16) * 4, 0, 0);
      }
    }

    // ---- S[16x64] = K·Qᵀ (pre-scaled) + edge_bias, 4 C-tiles ---------------
    v8f s[4];
#pragma unroll
    for (int ct = 0; ct < 4; ++ct) {
      v16h bq0 = load_b_frag(q16, KQ_DIM, j0 + ct * 16, 0,  lane);
      v16h bq1 = load_b_frag(q16, KQ_DIM, j0 + ct * 16, 32, lane);
      v8f c = {};
      c = __builtin_amdgcn_wmma_f32_16x16x32_f16(false, ak0, false, bq0, (short)0, c, false, false);
      c = __builtin_amdgcn_wmma_f32_16x16x32_f16(false, ak1, false, bq1, (short)0, c, false, false);
      const int jj = j0 + ct * 16 + nlo;
#pragma unroll
      for (int r = 0; r < 8; ++r) {
        const size_t eidx = ((size_t)(i0 + hgrp * 8 + r) * N_NODES + jj) * 4;
        const f32x4 e = __builtin_nontemporal_load((const f32x4*)(edge_attr + eidx));
        float bias = B2;
#pragma unroll
        for (int h = 0; h < 8; ++h) {           // relu(e@W1+b1)@W2 + b2
          float t = fmaf(e[0], W1[0][h], B1[h]);
          t = fmaf(e[1], W1[1][h], t);
          t = fmaf(e[2], W1[2][h], t);
          t = fmaf(e[3], W1[3][h], t);
          t = fmaxf(t, 0.0f);
          bias = fmaf(t, W2[h], bias);
        }
        c[r] += bias;
      }
      s[ct] = c;
    }

    // ---- online softmax over this 64-column step ---------------------------
#pragma unroll
    for (int r = 0; r < 8; ++r) {
      float tm = fmaxf(fmaxf(s[0][r], s[1][r]), fmaxf(s[2][r], s[3][r]));
#pragma unroll
      for (int m = 1; m < 16; m <<= 1) tm = fmaxf(tm, __shfl_xor(tm, m, 16));
      const float nm   = fmaxf(rmax[r], tm);
      const float corr = __expf(rmax[r] - nm);
      rmax[r] = nm;
      float ps = 0.0f;
#pragma unroll
      for (int ct = 0; ct < 4; ++ct) {
        const float p = __expf(s[ct][r] - nm);
        s[ct][r] = p;
        ps += p;
      }
#pragma unroll
      for (int m = 1; m < 16; m <<= 1) ps += __shfl_xor(ps, m, 16);
      rsum[r] = rsum[r] * corr + ps;
#pragma unroll
      for (int nt = 0; nt < 8; ++nt) acc[nt][r] *= corr;
    }

    // ---- stage P through LDS: C-layout -> [16][72] -> A fragments ----------
    _Float16* Pw = &Plds[wave][0];
#pragma unroll
    for (int ct = 0; ct < 4; ++ct)
#pragma unroll
      for (int r = 0; r < 8; ++r)
        Pw[(hgrp * 8 + r) * 72 + ct * 16 + nlo] = (_Float16)s[ct][r];
    asm volatile("" ::: "memory");  // keep LDS store->load order (HW keeps DS in-order)
    v16h ap0, ap1;
    {
      const _Float16* prow = Pw + (size_t)nlo * 72 + hgrp * 8;
      v8h l0 = *(const v8h*)(prow);
      v8h h0 = *(const v8h*)(prow + 16);
      v8h l1 = *(const v8h*)(prow + 32);
      v8h h1 = *(const v8h*)(prow + 48);
#pragma unroll
      for (int t = 0; t < 8; ++t) {
        ap0[t] = l0[t]; ap0[8 + t] = h0[t];
        ap1[t] = l1[t]; ap1[8 + t] = h1[t];
      }
    }

    // ---- acc += P[16x64] @ Vp[64x128] --------------------------------------
#pragma unroll
    for (int nt = 0; nt < 8; ++nt) {
      v16h bv0 = load_b_frag(vpT, N_NODES, nt * 16, j0,      lane);
      v16h bv1 = load_b_frag(vpT, N_NODES, nt * 16, j0 + 32, lane);
      acc[nt] = __builtin_amdgcn_wmma_f32_16x16x32_f16(false, ap0, false, bv0, (short)0, acc[nt], false, false);
      acc[nt] = __builtin_amdgcn_wmma_f32_16x16x32_f16(false, ap1, false, bv1, (short)0, acc[nt], false, false);
    }
  }

  // ---- merge the 8 waves' online-softmax partials ---------------------------
  if (nlo == 0) {
#pragma unroll
    for (int r = 0; r < 8; ++r) {
      stmax[wave][hgrp * 8 + r] = rmax[r];
      stsum[wave][hgrp * 8 + r] = rsum[r];
    }
  }
  __syncthreads();
#pragma unroll
  for (int r = 0; r < 8; ++r) {
    const int row = hgrp * 8 + r;
    float M = -__builtin_inff();
#pragma unroll
    for (int w = 0; w < 8; ++w) M = fmaxf(M, stmax[w][row]);
    float L = 0.0f;
#pragma unroll
    for (int w = 0; w < 8; ++w) L += stsum[w][row] * __expf(stmax[w][row] - M);
    const float f = __expf(rmax[r] - M) / L;
#pragma unroll
    for (int nt = 0; nt < 8; ++nt)
      atomicAdd(&outbuf[row][nt * 16 + nlo], acc[nt][r] * f);
  }
  __syncthreads();

  // ---- epilogue: v = x + attn@vp; out = v + relu(v@W1+b1)@W2 + b2 (WMMA) ----
  // reuse Plds: vt16 = rows of v as f16 [16][136]; hid16 likewise
  _Float16* vt16  = &Plds[0][0];  // 2176 halves, spans Plds[0..1]
  _Float16* hid16 = &Plds[2][0];  // 2176 halves, spans Plds[2..3]
  for (int idx = tid; idx < 16 * V_DIM; idx += 256) {
    const int row = idx >> 7, c = idx & 127;
    const float v = x[(size_t)(i0 + row) * IN_DIM + c] + outbuf[row][c];
    vtile[row][c] = v;
    vt16[row * 136 + c] = (_Float16)v;
  }
  __syncthreads();

  // layer 1: each wave computes its 16-column stripe of hidden
  {
    v8f c1 = {};
#pragma unroll
    for (int kf = 0; kf < 4; ++kf) {
      v16h a = load_a_frag(vt16, 136, kf * 32, lane);
      v16h b = load_b_frag(w1T16, IN_DIM, wave * 16, kf * 32, lane);
      c1 = __builtin_amdgcn_wmma_f32_16x16x32_f16(false, a, false, b, (short)0, c1, false, false);
    }
    const float b1v = l_b1[wave * 16 + nlo];
#pragma unroll
    for (int r = 0; r < 8; ++r)
      hid16[(hgrp * 8 + r) * 136 + wave * 16 + nlo] =
          (_Float16)fmaxf(c1[r] + b1v, 0.0f);
  }
  __syncthreads();

  // layer 2 + residual
  {
    v8f c2 = {};
#pragma unroll
    for (int kf = 0; kf < 4; ++kf) {
      v16h a = load_a_frag(hid16, 136, kf * 32, lane);
      v16h b = load_b_frag(w2T16, IN_DIM, wave * 16, kf * 32, lane);
      c2 = __builtin_amdgcn_wmma_f32_16x16x32_f16(false, a, false, b, (short)0, c2, false, false);
    }
    const float b2v = l_b2[wave * 16 + nlo];
#pragma unroll
    for (int r = 0; r < 8; ++r) {
      const int row = hgrp * 8 + r, col = wave * 16 + nlo;
      outbuf[row][col] = vtile[row][col] + c2[r] + b2v;
    }
  }
  __syncthreads();
  for (int idx = tid; idx < 16 * V_DIM; idx += 256)
    out[(size_t)(i0 + (idx >> 7)) * V_DIM + (idx & 127)] = outbuf[idx >> 7][idx & 127];
}

extern "C" void kernel_launch(void* const* d_in, const int* in_sizes, int n_in,
                              void* d_out, int out_size, void* d_ws, size_t ws_size,
                              hipStream_t stream) {
  (void)in_sizes; (void)n_in; (void)out_size; (void)ws_size;
  const float* x         = (const float*)d_in[0];
  const float* edge_attr = (const float*)d_in[1];
  const float* w_q  = (const float*)d_in[2];
  const float* w_k  = (const float*)d_in[3];
  const float* w_v  = (const float*)d_in[4];
  const float* b_v  = (const float*)d_in[5];
  const float* e_w1 = (const float*)d_in[6];
  const float* e_b1 = (const float*)d_in[7];
  const float* e_w2 = (const float*)d_in[8];
  const float* e_b2 = (const float*)d_in[9];
  const float* l_w1 = (const float*)d_in[10];
  const float* l_b1 = (const float*)d_in[11];
  const float* l_w2 = (const float*)d_in[12];
  const float* l_b2 = (const float*)d_in[13];
  float* out = (float*)d_out;

  _Float16* q16   = (_Float16*)d_ws;                     // [4096,64]
  _Float16* k16   = q16 + (size_t)N_NODES * KQ_DIM;      // [4096,64]
  _Float16* vpT   = k16 + (size_t)N_NODES * KQ_DIM;      // [128,4096]
  _Float16* w1T16 = vpT + (size_t)V_DIM * N_NODES;       // [128,128]
  _Float16* w2T16 = w1T16 + (size_t)IN_DIM * V_DIM;      // [128,128]

  prep_weights<<<(IN_DIM * V_DIM) / 256, 256, 0, stream>>>(l_w1, l_w2, w1T16, w2T16);
  proj_kernel<<<N_NODES / 16, 256, 0, stream>>>(x, w_q, w_k, w_v, b_v, q16, k16, vpT);
  attn_kernel<<<N_NODES / 16, 256, 0, stream>>>(x, edge_attr, q16, k16, vpT,
                                                w1T16, w2T16,
                                                e_w1, e_b1, e_w2, e_b2,
                                                l_b1, l_b2, out);
}
